// DMSpline_28166395527919
// MI455X (gfx1250) — matrix operand order repaired and checked by reading, not compiled, not run on hardware
//
#include <hip/hip_runtime.h>

typedef __attribute__((ext_vector_type(16))) _Float16 v16h;
typedef __attribute__((ext_vector_type(8)))  float    v8f;
typedef __attribute__((ext_vector_type(4)))  float    f32x4;

#define B_SZ   1024
#define C_SZ   20
#define N_SZ   26
#define VD     128
#define NSPL   10
#define NWORDS 100000

// ---------------------------------------------------------------------------
// Kernel 1: stream D -> d_out tail. 512MB, pure bandwidth. Non-temporal
// load+store (b128) so the 512MB stream does not evict the (hot, small)
// gather working set from the 192MB L2.
// ---------------------------------------------------------------------------
__global__ void dmspline_copyD(const f32x4* __restrict__ src,
                               f32x4* __restrict__ dst, size_t n4) {
    size_t i      = (size_t)blockIdx.x * blockDim.x + threadIdx.x;
    size_t stride = (size_t)gridDim.x * blockDim.x;
    for (; i < n4; i += stride) {
        f32x4 v = __builtin_nontemporal_load(&src[i]);
        __builtin_nontemporal_store(v, &dst[i]);
    }
}

// ---------------------------------------------------------------------------
// Kernel 2: one block (256 thr = 8 waves) per 16 batch rows.
// Phase 1 (all waves): x[r,:] = mean_c( D[knot[r],ctx[r,c],:]*xv[r] + W[ctx,:] )
//                      -> LDS as f16 (16 x 128 tile).
// Phase 2 (per wave, uniform n): diagonal-WMMA scores for its subset of n.
//   A rows  = x rows of the tile (16x128 f16)
//   B col j = O[:, tgt[tile*16+j, n]]   (gathered, f32->f16)
//   D = A*B : diag(i,i) = scores[tile*16+i, n]; K=128 via 4 chained
//   v_wmma_f32_16x16x32_f16.
// ---------------------------------------------------------------------------
__global__ void dmspline_scores(const float* __restrict__ x_vals,
                                const int*   __restrict__ knot_ids,
                                const int*   __restrict__ ctx_ids,
                                const int*   __restrict__ tgt_ids,
                                const float* __restrict__ D,
                                const float* __restrict__ W,
                                const float* __restrict__ O,
                                float* __restrict__ scores) {
    __shared__ _Float16 xs[16][VD];
    __shared__ int   ctx_s[16 * C_SZ];
    __shared__ int   knot_s[16];
    __shared__ float xv_s[16];

    const int tid  = threadIdx.x;
    const int row0 = blockIdx.x * 16;

    if (tid < 16) {
        knot_s[tid] = knot_ids[row0 + tid];
        xv_s[tid]   = x_vals[row0 + tid];
    }
    for (int e = tid; e < 16 * C_SZ; e += blockDim.x)
        ctx_s[e] = ctx_ids[(row0 + e / C_SZ) * C_SZ + (e % C_SZ)];
    __syncthreads();

    // ---- Phase 1: build f16 x-tile in LDS (coalesced row reads of D/W) ----
    for (int e = tid; e < 16 * VD; e += blockDim.x) {
        const int r = e >> 7;            // row in tile
        const int v = e & (VD - 1);      // dim
        float accD = 0.f, accW = 0.f;
        const size_t dbase = (size_t)knot_s[r] * (size_t)NWORDS;
        #pragma unroll 4
        for (int c = 0; c < C_SZ; ++c) {
            const int w = ctx_s[r * C_SZ + c];
            accD += D[(dbase + (size_t)w) * VD + v];
            accW += W[(size_t)w * VD + v];
        }
        xs[r][v] = (_Float16)((accD * xv_s[r] + accW) * (1.0f / C_SZ));
    }
    __syncthreads();

    // ---- Phase 2: WMMA diagonal scores ----
    const int lane = tid & 31;
    const int wave = tid >> 5;
    const int arow = lane & 15;          // A: row of x tile held by this lane
    const int hi   = lane >> 4;          // upper half-wave selector
    const int bcol = lane & 15;          // B: output column (== diag row)

    for (int n = wave; n < N_SZ; n += 8) {             // wave-uniform => EXEC all 1s
        const int w = tgt_ids[(row0 + bcol) * N_SZ + n]; // O column for B col `bcol`
        v8f acc = {};
        #pragma unroll
        for (int kk = 0; kk < 4; ++kk) {
            const int k0 = kk * 32;
            v16h a, b;
            #pragma unroll
            for (int t = 0; t < 8; ++t) {              // A 16x32 f16 layout
                a[t]     = xs[arow][k0 + hi * 8 + t];
                a[8 + t] = xs[arow][k0 + 16 + hi * 8 + t];
            }
            #pragma unroll
            for (int t = 0; t < 16; ++t) {             // B 32x16 f16 layout
                b[t] = (_Float16)O[(size_t)(k0 + (t < 8 ? hi * 8 + t : 16 + hi * 8 + (t - 8))) * (size_t)NWORDS
                                   + (size_t)w];
            }
            acc = __builtin_amdgcn_wmma_f32_16x16x32_f16(
                      false, a, false, b, (short)0, acc, false, false);
        }
        // C/D layout: diag (i,i): i<8 -> lane i, elem i ; i>=8 -> lane i+16, elem i-8
        union { v8f v; float f[8]; } u; u.v = acc;
        if (lane < 8) {
            scores[(size_t)(row0 + lane) * N_SZ + n] = u.f[lane];
        } else if (lane >= 24) {
            const int i = lane - 16;
            scores[(size_t)(row0 + i) * N_SZ + n] = u.f[lane - 24];
        }
    }
}

extern "C" void kernel_launch(void* const* d_in, const int* in_sizes, int n_in,
                              void* d_out, int out_size, void* d_ws, size_t ws_size,
                              hipStream_t stream) {
    const float* x_vals   = (const float*)d_in[0];
    const int*   knot_ids = (const int*)  d_in[1];
    const int*   ctx_ids  = (const int*)  d_in[2];
    const int*   tgt_ids  = (const int*)  d_in[3];
    const float* D        = (const float*)d_in[4];
    const float* W        = (const float*)d_in[5];
    const float* O        = (const float*)d_in[6];

    float* scores = (float*)d_out;
    float* D_out  = scores + (size_t)B_SZ * N_SZ;   // byte off 106496 (16B aligned)

    const size_t n4 = ((size_t)NSPL * NWORDS * VD) / 4;   // 32M float4
    dmspline_copyD<<<4096, 256, 0, stream>>>((const f32x4*)D, (f32x4*)D_out, n4);
    dmspline_scores<<<B_SZ / 16, 256, 0, stream>>>(x_vals, knot_ids, ctx_ids,
                                                   tgt_ids, D, W, O, scores);
}
